// Model_3281355014319
// MI455X (gfx1250) — matrix operand order repaired and checked by reading, not compiled, not run on hardware
//
#include <hip/hip_runtime.h>
#include <hip/hip_bf16.h>
#include <math.h>

typedef __attribute__((ext_vector_type(16))) __bf16 v16bf;
typedef __attribute__((ext_vector_type(8)))  float  v8f;
typedef __attribute__((__vector_size__(4 * sizeof(int)))) int v4i;

constexpr int BN = 16;     // batch
constexpr int TT = 750;    // time
constexpr int DD = 2048;   // feature dim
constexpr int CC = 21;     // classes + bg
constexpr int KK = 8;      // clusters
constexpr int KACT = 93;   // 750/8
constexpr int TP = 752;    // padded time (zero row front+back)
constexpr int KD = 3 * DD; // im2col K = 6144

// ---- output offsets (floats, reference tuple order) ----
constexpr long long OFF_VS    = 0;
constexpr long long OFF_RVS   = 320;
constexpr long long OFF_CASF  = 640;
constexpr long long OFF_RCASF = OFF_CASF + (long long)BN * TT * CC;
constexpr long long OFF_FEAT  = OFF_RCASF + (long long)BN * TT * CC;
constexpr long long OFF_OCA   = OFF_FEAT + (long long)BN * TT * DD;
constexpr long long OFF_MCA   = OFF_OCA + BN * CC;
constexpr long long OFF_OCW   = OFF_MCA + BN * CC;
constexpr long long OFF_MCW   = OFF_OCW + BN * CC;
constexpr long long OFF_OATT  = OFF_MCW + BN * CC;
constexpr long long OFF_MATT  = OFF_OATT + BN * TT;
constexpr long long OFF_OFRM  = OFF_MATT + BN * TT;
constexpr long long OFF_MFRM  = OFF_OFRM + (long long)BN * TT * CC;

static __device__ __forceinline__ float wave_red(float v) {
  #pragma unroll
  for (int o = 16; o; o >>= 1) v += __shfl_xor(v, o, 32);
  return v;
}
static __device__ __forceinline__ float sigm(float x) { return 1.f / (1.f + expf(-x)); }

// ---- gfx1250 async global->LDS staging (guarded; fallback = sync copy) ----
#if defined(__has_builtin)
#if __has_builtin(__builtin_amdgcn_global_load_async_to_lds_b128)
#define ASYNC_LDS 1
#endif
#endif
#ifndef ASYNC_LDS
#define ASYNC_LDS 0
#endif

static __device__ __forceinline__ void wait_async0() {
#if ASYNC_LDS
#if defined(__has_builtin) && __has_builtin(__builtin_amdgcn_s_wait_asynccnt)
  __builtin_amdgcn_s_wait_asynccnt(0);
#else
  asm volatile("s_wait_asynccnt 0x0" ::: "memory");
#endif
#endif
}

// ============================================================
// Prep: pad x (B,T,D)->(B,752,D) bf16 with zero rows at t=0,751
// ============================================================
__global__ void k_pad_x(const float* __restrict__ x, __bf16* __restrict__ xp, int total) {
  int idx = blockIdx.x * 256 + threadIdx.x;
  if (idx >= total) return;
  int d = idx % DD;
  int rem = idx / DD;
  int t = rem % TP;
  int b = rem / TP;
  float v = 0.f;
  if (t >= 1 && t <= TT) v = x[((size_t)b * TT + (t - 1)) * DD + d];
  xp[idx] = (__bf16)v;
}

// Permute conv weight (O,I,3) -> wb[o][s*D+i] bf16 (contiguous-K rows for B frags)
__global__ void k_conv_w(const float* __restrict__ w, __bf16* __restrict__ wb, int total) {
  int idx = blockIdx.x * 256 + threadIdx.x;
  if (idx >= total) return;
  int kp = idx % KD;
  int o  = idx / KD;
  int s  = kp / DD;
  int i  = kp % DD;
  wb[idx] = (__bf16)w[((size_t)o * DD + i) * 3 + s];
}

// ============================================================
// Conv1d as GEMM (12000 x 2048 x 6144), cooperative LDS pipeline.
// Workgroup (8 waves) tile 96x256; per 32-wide K step:
//   stage A(96x32)+B(256x32)=22KB into LDS (async-to-LDS, double buffered),
//   each wave computes 48x64 via 14 ds_load_b128 + 12 v_wmma_f32_16x16x32_bf16.
// ============================================================
union AF { v16bf v; uint4 q[2]; };

__global__ __launch_bounds__(256) void k_conv_wmma(const __bf16* __restrict__ xp,
                                                   const __bf16* __restrict__ wb,
                                                   const float* __restrict__ bias,
                                                   float* __restrict__ out) {
  __shared__ alignas(16) __bf16 sA[2][96 * 32];
  __shared__ alignas(16) __bf16 sB[2][256 * 32];

  const int tid  = threadIdx.x;
  const int lane = tid & 31;
  const int wv   = tid >> 5;   // wave 0..7
  const int wy   = wv >> 2;    // 0..1 -> M sub-tile
  const int wx   = wv & 3;     // 0..3 -> N sub-tile
  const int lr   = lane & 15;
  const int half = lane >> 4;
  const int nb = blockIdx.x * 256;
  const int mb = blockIdx.y * 96;

  // ---- staging assignment: thread -> 2 A chunks + 4 B chunks (16B each) ----
  const int seg = tid & 3;        // 16B segment within a 64B (32-elem) row
  const int ra0 = tid >> 2;       // A row 0..63
  const int ra1 = ra0 + 64;       // A row 64..127
  const __bf16 *gA0, *gA1;
  {
    int r = mb + ra0; gA0 = xp + ((size_t)(r / TT) * TP + (r % TT)) * DD + seg * 8;
    r = mb + ra1;     gA1 = xp + ((size_t)(r / TT) * TP + (r % TT)) * DD + seg * 8;
  }
  const __bf16* gB[4];
  #pragma unroll
  for (int j = 0; j < 4; ++j) gB[j] = wb + (size_t)(nb + (tid >> 2) + j * 64) * KD + seg * 8;
  const int aoff0 = ra0 * 32 + seg * 8;
  const int aoff1 = ra1 * 32 + seg * 8;
  int boff[4];
  #pragma unroll
  for (int j = 0; j < 4; ++j) boff[j] = ((tid >> 2) + j * 64) * 32 + seg * 8;

  auto stage = [&](int bi, int kb) {
#if ASYNC_LDS
    __builtin_amdgcn_global_load_async_to_lds_b128((v4i*)(gA0 + kb), (v4i*)&sA[bi][aoff0], 0, 0);
    __builtin_amdgcn_global_load_async_to_lds_b128((v4i*)(gA1 + kb), (v4i*)&sA[bi][aoff1], 0, 0);
    #pragma unroll
    for (int j = 0; j < 4; ++j)
      __builtin_amdgcn_global_load_async_to_lds_b128((v4i*)(gB[j] + kb), (v4i*)&sB[bi][boff[j]], 0, 0);
#else
    *(uint4*)&sA[bi][aoff0] = *(const uint4*)(gA0 + kb);
    *(uint4*)&sA[bi][aoff1] = *(const uint4*)(gA1 + kb);
    #pragma unroll
    for (int j = 0; j < 4; ++j)
      *(uint4*)&sB[bi][boff[j]] = *(const uint4*)(gB[j] + kb);
#endif
  };

  v8f acc[3][4];
  #pragma unroll
  for (int i = 0; i < 3; ++i)
    #pragma unroll
    for (int j = 0; j < 4; ++j) acc[i][j] = {};

  stage(0, 0);
  wait_async0();
  __syncthreads();

  int buf = 0;
  for (int kb = 0; kb < KD; kb += 32) {
    if (kb + 32 < KD) stage(buf ^ 1, kb + 32);

    const __bf16* As = &sA[buf][0];
    const __bf16* Bs = &sB[buf][0];
    AF a[3], bf[4];
    // A 16x32 bf16 frag: lane<16 -> K {0..7,16..23}; lane>=16 -> +8
    #pragma unroll
    for (int i = 0; i < 3; ++i) {
      const int ro = (wy * 48 + i * 16 + lr) * 32;
      a[i].q[0] = *(const uint4*)(As + ro + half * 8);
      a[i].q[1] = *(const uint4*)(As + ro + 16 + half * 8);
    }
    // B 32x16 bf16 frag: lane<16 -> K {0..7,16..23}; lane>=16 -> +8
    #pragma unroll
    for (int j = 0; j < 4; ++j) {
      const int ro = (wx * 64 + j * 16 + lr) * 32;
      bf[j].q[0] = *(const uint4*)(Bs + ro + half * 8);
      bf[j].q[1] = *(const uint4*)(Bs + ro + 16 + half * 8);
    }
    #pragma unroll
    for (int i = 0; i < 3; ++i)
      #pragma unroll
      for (int j = 0; j < 4; ++j)
        acc[i][j] = __builtin_amdgcn_wmma_f32_16x16x32_bf16(false, a[i].v, false, bf[j].v,
                                                            (short)0, acc[i][j], false, false);
    wait_async0();
    __syncthreads();
    buf ^= 1;
  }

  // Epilogue: D layout — lane<16: M=e, N=lane; lane>=16: M=8+e, N=lane-16
  #pragma unroll
  for (int j = 0; j < 4; ++j) {
    const int n = nb + wx * 64 + j * 16 + lr;
    const float bn = bias[n];
    #pragma unroll
    for (int i = 0; i < 3; ++i) {
      const int mrow = mb + wy * 48 + i * 16 + half * 8;
      #pragma unroll
      for (int e = 0; e < 8; ++e) {
        float v = acc[i][j][e] + bn;
        out[(size_t)(mrow + e) * DD + n] = v > 0.f ? v : 0.f;
      }
    }
  }
}

// ============================================================
// EM / random-walk assignment: z[b,k,t] = softmax_k(5 * l1n(mu_k)·l1n(X_t))
// ============================================================
__global__ __launch_bounds__(32) void k_em_z(const float* __restrict__ X,
                                             const float* __restrict__ mu,
                                             float* __restrict__ z,
                                             float* __restrict__ zsum) {
  const int lane = threadIdx.x;
  const int row = blockIdx.x;
  const int b = row / TT, t = row % TT;
  const float* xr = X + (size_t)row * DD;
  const float* mb = mu + (size_t)b * KK * DD;
  float dot[KK] = {}, ms[KK] = {}, xs = 0.f;
  for (int d = lane; d < DD; d += 32) {
    float xv = xr[d];
    xs += xv * xv;
    #pragma unroll
    for (int k = 0; k < KK; ++k) {
      float mv = mb[k * DD + d];
      dot[k] += mv * xv;
      ms[k] += mv * mv;
    }
  }
  xs = wave_red(xs);
  #pragma unroll
  for (int k = 0; k < KK; ++k) { dot[k] = wave_red(dot[k]); ms[k] = wave_red(ms[k]); }
  float inx = 1.f / (sqrtf(xs) + 1e-9f);
  float sc[KK], mx = -1e30f;
  #pragma unroll
  for (int k = 0; k < KK; ++k) {
    sc[k] = 5.f * dot[k] * inx / (sqrtf(ms[k]) + 1e-9f);
    mx = fmaxf(mx, sc[k]);
  }
  float se = 0.f;
  #pragma unroll
  for (int k = 0; k < KK; ++k) { sc[k] = expf(sc[k] - mx); se += sc[k]; }
  float ise = 1.f / se;
  if (lane == 0) {
    #pragma unroll
    for (int k = 0; k < KK; ++k) {
      float zv = sc[k] * ise;
      z[((size_t)b * KK + k) * TT + t] = zv;
      if (zsum) atomicAdd(&zsum[b * KK + k], zv);
    }
  }
}

// mu[b,k,d] = sum_t (z/(zsum+1e-9)) * x[b,t,d]
__global__ __launch_bounds__(256) void k_em_mu(const float* __restrict__ X,
                                               const float* __restrict__ z,
                                               const float* __restrict__ zsum,
                                               float* __restrict__ mu) {
  int d = blockIdx.x * 256 + threadIdx.x;
  int k = blockIdx.y, b = blockIdx.z;
  const float* zr = z + ((size_t)b * KK + k) * TT;
  float inv = 1.f / (zsum[b * KK + k] + 1e-9f);
  float acc = 0.f;
  for (int t = 0; t < TT; ++t) acc += zr[t] * X[((size_t)b * TT + t) * DD + d];
  mu[((size_t)b * KK + k) * DD + d] = acc * inv;
}

__global__ void k_mu_init(const float* __restrict__ mp, float* __restrict__ mu, int total) {
  int idx = blockIdx.x * 256 + threadIdx.x;
  if (idx >= total) return;
  mu[idx] = mp[idx % (KK * DD)];
}

__global__ void k_zero(float* __restrict__ p, int n) {
  int i = blockIdx.x * 256 + threadIdx.x;
  if (i < n) p[i] = 0.f;
}

// inv norms of ac_center rows (0..20) and fg_center (21)
__global__ __launch_bounds__(32) void k_acnorm(const float* __restrict__ ac,
                                               const float* __restrict__ fg,
                                               float* __restrict__ invn) {
  int row = blockIdx.x, lane = threadIdx.x;
  const float* p = (row < CC) ? (ac + (size_t)row * DD) : fg;
  float s = 0.f;
  for (int d = lane; d < DD; d += 32) { float v = p[d]; s += v * v; }
  s = wave_red(s);
  if (lane == 0) invn[row] = 1.f / (sqrtf(s) + 1e-9f);
}

// ============================================================
// Per-frame mega kernel: one wave per (b,t).
// sums layout: [0..15] att_o, [16..31] att_m, [32..367] sf_o, [368..703] sf_m
// ============================================================
__global__ __launch_bounds__(32) void k_frame(const float* __restrict__ feat,
                                              const float* __restrict__ mu,
                                              const float* __restrict__ z2,
                                              const float* __restrict__ cls,
                                              const float* __restrict__ ac,
                                              const float* __restrict__ fg,
                                              const float* __restrict__ invn,
                                              float* __restrict__ out,
                                              float* __restrict__ sfo, float* __restrict__ sfm,
                                              float* __restrict__ csb, float* __restrict__ rcsb,
                                              float* __restrict__ sums) {
  const int lane = threadIdx.x;
  const int row = blockIdx.x;
  const int b = row / TT, t = row % TT;
  const float* fr = feat + (size_t)row * DD;
  const float* mb = mu + (size_t)b * KK * DD;
  float zk[KK];
  #pragma unroll
  for (int k = 0; k < KK; ++k) zk[k] = 0.25f * z2[((size_t)b * KK + k) * TT + t];

  float cas[CC] = {}, rcas[CC] = {}, frm[CC] = {}, mfrm[CC] = {};
  float fgf = 0.f, fgr = 0.f, fs = 0.f, rs = 0.f;
  for (int d = lane; d < DD; d += 32) {
    float f = fr[d];
    float s = 0.f;
    #pragma unroll
    for (int k = 0; k < KK; ++k) s += zk[k] * mb[k * DD + d];
    float r = 0.5f * f + s;
    fs += f * f; rs += r * r;
    float g = fg[d]; fgf += f * g; fgr += r * g;
    #pragma unroll
    for (int c = 0; c < CC; ++c) {
      float cw = cls[c * DD + d]; cas[c] += f * cw; rcas[c] += r * cw;
      float av = ac[c * DD + d];  frm[c] += f * av; mfrm[c] += r * av;
    }
  }
  fs = wave_red(fs); rs = wave_red(rs); fgf = wave_red(fgf); fgr = wave_red(fgr);
  #pragma unroll
  for (int c = 0; c < CC; ++c) {
    cas[c] = wave_red(cas[c]); rcas[c] = wave_red(rcas[c]);
    frm[c] = wave_red(frm[c]); mfrm[c] = wave_red(mfrm[c]);
  }
  if (lane != 0) return;
  float invf = 1.f / (sqrtf(fs) + 1e-9f);
  float invr = 1.f / (sqrtf(rs) + 1e-9f);
  float ao = sigm(20.f * fgf * invf * invn[CC]);
  float am = sigm(20.f * fgr * invr * invn[CC]);
  out[OFF_OATT + row] = ao;
  out[OFF_MATT + row] = am;
  atomicAdd(&sums[b], ao);
  atomicAdd(&sums[BN + b], am);
  float cs20 = sigm(cas[CC - 1]);
  float rcs20 = sigm(rcas[CC - 1]);
  #pragma unroll
  for (int c = 0; c < CC; ++c) {
    float of = 20.f * frm[c] * invf * invn[c];
    float mf = 20.f * mfrm[c] * invr * invn[c];
    out[OFF_OFRM + (size_t)row * CC + c] = of;
    out[OFF_MFRM + (size_t)row * CC + c] = mf;
    float so = sigm(of), sm = sigm(mf);
    sfo[(size_t)row * CC + c] = so;
    sfm[(size_t)row * CC + c] = sm;
    atomicAdd(&sums[32 + b * CC + c], so);
    atomicAdd(&sums[32 + BN * CC + b * CC + c], sm);
    float csv = sigm(cas[c]), rcv = sigm(rcas[c]);
    csb[(size_t)row * CC + c] = csv;
    rcsb[(size_t)row * CC + c] = rcv;
    out[OFF_CASF  + (size_t)row * CC + c] = (c < CC - 1) ? csv * (1.f - cs20)  : cs20;
    out[OFF_RCASF + (size_t)row * CC + c] = (c < CC - 1) ? rcv * (1.f - rcs20) : rcs20;
  }
}

// ============================================================
// Top-k(93)/750 mean per (b, c<20): in-LDS bitonic sort (descending)
// ============================================================
__global__ __launch_bounds__(256) void k_topk(const float* __restrict__ csb,
                                              const float* __restrict__ rcsb,
                                              float* __restrict__ out) {
  __shared__ float sh[1024];
  const int c = blockIdx.x, b = blockIdx.y, which = blockIdx.z;
  const float* s = which ? rcsb : csb;
  const int tid = threadIdx.x;
  for (int i = tid; i < 1024; i += 256)
    sh[i] = (i < TT) ? s[((size_t)b * TT + i) * CC + c] : -1e30f;
  for (int k = 2; k <= 1024; k <<= 1) {
    for (int j = k >> 1; j > 0; j >>= 1) {
      __syncthreads();
      for (int i = tid; i < 1024; i += 256) {
        int ixj = i ^ j;
        if (ixj > i) {
          float a = sh[i], bb = sh[ixj];
          bool desc = ((i & k) == 0);
          if (desc ? (a < bb) : (a > bb)) { sh[i] = bb; sh[ixj] = a; }
        }
      }
    }
  }
  __syncthreads();
  if (tid == 0) {
    float acc = 0.f;
    for (int i = 0; i < KACT; ++i) acc += sh[i];
    out[(which ? OFF_RVS : OFF_VS) + b * 20 + c] = acc / (float)KACT;
  }
}

// ============================================================
// Video-level weighted features (ca + cw, o + m variants fused)
// ============================================================
__global__ __launch_bounds__(256) void k_vidfeat(const float* __restrict__ feat,
                                                 const float* __restrict__ mu,
                                                 const float* __restrict__ z2,
                                                 const float* __restrict__ atto,
                                                 const float* __restrict__ attm,
                                                 const float* __restrict__ sfo,
                                                 const float* __restrict__ sfm,
                                                 const float* __restrict__ sums,
                                                 float* __restrict__ cafo, float* __restrict__ cafm,
                                                 float* __restrict__ cwfo, float* __restrict__ cwfm) {
  const int tid = threadIdx.x;
  const int b = blockIdx.y;
  const int d = blockIdx.x * 256 + tid;
  __shared__ float s_ao[64], s_am[64], s_so[64 * CC], s_sm[64 * CC], s_z[64 * KK];
  float muk[KK];
  #pragma unroll
  for (int k = 0; k < KK; ++k) muk[k] = mu[((size_t)b * KK + k) * DD + d];
  float cao = 0.f, cam = 0.f, cwo[CC] = {}, cwm[CC] = {};
  for (int t0 = 0; t0 < TT; t0 += 64) {
    int nT = (TT - t0 < 64) ? (TT - t0) : 64;
    __syncthreads();
    for (int i = tid; i < nT; i += 256) {
      s_ao[i] = atto[b * TT + t0 + i];
      s_am[i] = attm[b * TT + t0 + i];
    }
    for (int i = tid; i < nT * CC; i += 256) {
      int t_ = i / CC, c_ = i % CC;
      s_so[i] = sfo[((size_t)b * TT + t0 + t_) * CC + c_];
      s_sm[i] = sfm[((size_t)b * TT + t0 + t_) * CC + c_];
    }
    for (int i = tid; i < nT * KK; i += 256) {
      int t_ = i / KK, k_ = i % KK;
      s_z[i] = z2[((size_t)b * KK + k_) * TT + t0 + t_];
    }
    __syncthreads();
    for (int tt = 0; tt < nT; ++tt) {
      float f = feat[((size_t)b * TT + t0 + tt) * DD + d];
      float s = 0.f;
      #pragma unroll
      for (int k = 0; k < KK; ++k) s += s_z[tt * KK + k] * muk[k];
      float r = 0.5f * f + 0.25f * s;
      cao += f * s_ao[tt];
      cam += r * s_am[tt];
      #pragma unroll
      for (int c = 0; c < CC; ++c) { cwo[c] += f * s_so[tt * CC + c]; cwm[c] += r * s_sm[tt * CC + c]; }
    }
  }
  cafo[(size_t)b * DD + d] = cao * (1.f / (sums[b] + 1e-5f));
  cafm[(size_t)b * DD + d] = cam * (1.f / (sums[BN + b] + 1e-5f));
  #pragma unroll
  for (int c = 0; c < CC; ++c) {
    cwfo[((size_t)b * CC + c) * DD + d] = cwo[c] * (1.f / (sums[32 + b * CC + c] + 1e-5f));
    cwfm[((size_t)b * CC + c) * DD + d] = cwm[c] * (1.f / (sums[32 + BN * CC + b * CC + c] + 1e-5f));
  }
}

// raw video scores: sca[b,c]=20*l1n(caf_b)·l1n(ac_c), scw[b,c]=20*l1n(cwf_bc)·l1n(ac_c)
__global__ __launch_bounds__(32) void k_vidscr(const float* __restrict__ caf,
                                               const float* __restrict__ cwf,
                                               const float* __restrict__ ac,
                                               const float* __restrict__ invn,
                                               float* __restrict__ sca, float* __restrict__ scw) {
  const int c = blockIdx.x, b = blockIdx.y, lane = threadIdx.x;
  const float* acr = ac + (size_t)c * DD;
  const float* cwr = cwf + ((size_t)b * CC + c) * DD;
  const float* car = caf + (size_t)b * DD;
  float d1 = 0, s1 = 0, d2 = 0, s2 = 0;
  for (int d = lane; d < DD; d += 32) {
    float a = acr[d], v = cwr[d], u = car[d];
    d1 += v * a; s1 += v * v; d2 += u * a; s2 += u * u;
  }
  d1 = wave_red(d1); s1 = wave_red(s1); d2 = wave_red(d2); s2 = wave_red(s2);
  if (lane == 0) {
    scw[b * CC + c] = 20.f * d1 * invn[c] / (sqrtf(s1) + 1e-9f);
    sca[b * CC + c] = 20.f * d2 * invn[c] / (sqrtf(s2) + 1e-9f);
  }
}

__global__ __launch_bounds__(32) void k_softmax21(const float* __restrict__ s0, const float* __restrict__ s1,
                                                  const float* __restrict__ s2, const float* __restrict__ s3,
                                                  float* __restrict__ d0, float* __restrict__ d1,
                                                  float* __restrict__ d2, float* __restrict__ d3) {
  int arr = blockIdx.x, b = blockIdx.y, lane = threadIdx.x;
  const float* s = arr == 0 ? s0 : arr == 1 ? s1 : arr == 2 ? s2 : s3;
  float* d = arr == 0 ? d0 : arr == 1 ? d1 : arr == 2 ? d2 : d3;
  float v = (lane < CC) ? s[b * CC + lane] : -1e30f;
  float m = v;
  #pragma unroll
  for (int o = 16; o; o >>= 1) m = fmaxf(m, __shfl_xor(m, o, 32));
  float e = (lane < CC) ? expf(v - m) : 0.f;
  float se = wave_red(e);
  if (lane < CC) d[b * CC + lane] = e / se;
}

// ============================================================
extern "C" void kernel_launch(void* const* d_in, const int* in_sizes, int n_in,
                              void* d_out, int out_size, void* d_ws, size_t ws_size,
                              hipStream_t stream) {
  const float* x      = (const float*)d_in[0];
  const float* conv_w = (const float*)d_in[1];
  const float* conv_b = (const float*)d_in[2];
  const float* cls_w  = (const float*)d_in[3];
  const float* mu_p   = (const float*)d_in[4];
  const float* ac     = (const float*)d_in[5];
  const float* fg     = (const float*)d_in[6];
  float* out = (float*)d_out;

  char* w = (char*)d_ws;
  auto carve = [&](size_t bytes) -> char* {
    char* p = w;
    w += (bytes + 255) & ~(size_t)255;
    return p;
  };
  __bf16* xp  = (__bf16*)carve((size_t)BN * TP * DD * 2);
  __bf16* wbw = (__bf16*)carve((size_t)DD * KD * 2);
  float* z    = (float*)carve((size_t)BN * KK * TT * 4);
  float* z2   = (float*)carve((size_t)BN * KK * TT * 4);
  float* mu0  = (float*)carve((size_t)BN * KK * DD * 4);
  float* mu1  = (float*)carve((size_t)BN * KK * DD * 4);
  float* zsum = (float*)carve(512);
  float* invn = (float*)carve(256);
  float* sfo  = (float*)carve((size_t)BN * TT * CC * 4);
  float* sfm  = (float*)carve((size_t)BN * TT * CC * 4);
  float* csb  = (float*)carve((size_t)BN * TT * CC * 4);
  float* rcsb = (float*)carve((size_t)BN * TT * CC * 4);
  float* sums = (float*)carve(4096);
  float* cafo = (float*)carve((size_t)BN * DD * 4);
  float* cafm = (float*)carve((size_t)BN * DD * 4);
  float* cwfo = (float*)carve((size_t)BN * CC * DD * 4);
  float* cwfm = (float*)carve((size_t)BN * CC * DD * 4);
  float* scr  = (float*)carve(8192);
  float *sca_o = scr, *scw_o = scr + BN * CC, *sca_m = scr + 2 * BN * CC, *scw_m = scr + 3 * BN * CC;

  float* feat = out + OFF_FEAT;

  { int tot = BN * TP * DD; k_pad_x<<<(tot + 255) / 256, 256, 0, stream>>>(x, xp, tot); }
  { int tot = DD * KD;      k_conv_w<<<(tot + 255) / 256, 256, 0, stream>>>(conv_w, wbw, tot); }

  // conv as WMMA GEMM -> features (bias + relu) straight into d_out
  // workgroup tile 96x256: grid = (2048/256, 12000/96) = (8, 125), exact.
  k_conv_wmma<<<dim3(DD / 256, (BN * TT) / 96), 256, 0, stream>>>(xp, wbw, conv_b, feat);

  // EM soft-clustering, 2 iterations
  { int tot = BN * KK * DD; k_mu_init<<<(tot + 255) / 256, 256, 0, stream>>>(mu_p, mu0, tot); }
  k_zero<<<1, 256, 0, stream>>>(zsum, BN * KK);
  k_em_z<<<BN * TT, 32, 0, stream>>>(x, mu0, z, zsum);
  k_em_mu<<<dim3(DD / 256, KK, BN), 256, 0, stream>>>(x, z, zsum, mu1);
  k_zero<<<1, 256, 0, stream>>>(zsum, BN * KK);
  k_em_z<<<BN * TT, 32, 0, stream>>>(x, mu1, z, zsum);
  k_em_mu<<<dim3(DD / 256, KK, BN), 256, 0, stream>>>(x, z, zsum, mu0);

  // random-walk assignment on features with final mu
  k_em_z<<<BN * TT, 32, 0, stream>>>(feat, mu0, z2, nullptr);

  // prediction-module constants + accumulators
  k_acnorm<<<CC + 1, 32, 0, stream>>>(ac, fg, invn);
  k_zero<<<3, 256, 0, stream>>>(sums, 32 + 2 * BN * CC);

  // per-frame fused pass (cas/r_cas/frm/m_frm/att + fuse + sigmoid caches)
  k_frame<<<BN * TT, 32, 0, stream>>>(feat, mu0, z2, cls_w, ac, fg, invn,
                                      out, sfo, sfm, csb, rcsb, sums);

  // vs / r_vs (top-93 mean per (b, c<20))
  k_topk<<<dim3(20, BN, 2), 256, 0, stream>>>(csb, rcsb, out);

  // video-level features then scores then softmax
  k_vidfeat<<<dim3(DD / 256, BN), 256, 0, stream>>>(feat, mu0, z2,
                                                    out + OFF_OATT, out + OFF_MATT,
                                                    sfo, sfm, sums, cafo, cafm, cwfo, cwfm);
  k_vidscr<<<dim3(CC, BN), 32, 0, stream>>>(cafo, cwfo, ac, invn, sca_o, scw_o);
  k_vidscr<<<dim3(CC, BN), 32, 0, stream>>>(cafm, cwfm, ac, invn, sca_m, scw_m);
  k_softmax21<<<dim3(4, BN), 32, 0, stream>>>(sca_o, sca_m, scw_o, scw_m,
                                              out + OFF_OCA, out + OFF_MCA,
                                              out + OFF_OCW, out + OFF_MCW);
}
